// GATLayer_37615323578852
// MI455X (gfx1250) — matrix operand order repaired and checked by reading, not compiled, not run on hardware
//
#include <hip/hip_runtime.h>
#include <hip/hip_bf16.h>
#include <math.h>

// ---------------------------------------------------------------------------
// GAT layer, sparse formulation:
//   H = X @ W                       (bf16 WMMA GEMM, f32 accum)
//   s1[i] = H[i] . a[0:256] ; s2[i] = H[i] . a[256:512]
//   e_k = leakyrelu(s1[src_k] + s2[tgt_k])
//   per-src-row softmax over edges (max via f32 atomic max, sum via atomic add)
//   out[src] += (w_k / rowsum[src]) * H[tgt]  ;  + bias
//   rows with no out-edges -> uniform softmax -> mean(H) + bias
// ---------------------------------------------------------------------------

#define D 256          // IN_DIM == OUT_DIM == 256
#define ALPHA 0.2f
#define AP 264         // padded LDS row stride (bf16 elems): 528 B, 16B-aligned

typedef __attribute__((ext_vector_type(16))) __bf16 v16bf;
typedef __attribute__((ext_vector_type(8)))  float  v8f;

union Frag16 {           // 32-byte WMMA operand assembled from two 16B loads
  uint4  q[2];
  v16bf  v;
};
union Pack4 {            // 4 bf16 <-> uint2 for vectorized LDS stores
  __bf16 h[4];
  uint2  u;
};

// ------------------------- init small buffers ------------------------------
__global__ void init_buffers(float* __restrict__ rowmax, float* __restrict__ rowsum,
                             float* __restrict__ meanacc, int n) {
  int idx = blockIdx.x * 256 + threadIdx.x;
  if (idx < n) { rowmax[idx] = -3.0e38f; rowsum[idx] = 0.0f; }
  if (idx < D) meanacc[idx] = 0.0f;
}

// ------------------------- W -> bf16, transposed ---------------------------
// Wt[c*D + k] = bf16(W[k*D + c]); lane's B fragment becomes 32 contiguous bytes.
__global__ void prep_w(const float* __restrict__ W, __bf16* __restrict__ Wt) {
  int k = blockIdx.x;     // 0..255
  int c = threadIdx.x;    // 0..255 (coalesced read of row k)
  Wt[(size_t)c * D + k] = (__bf16)W[(size_t)k * D + c];
}

// ------------------------- H = X @ W  (WMMA bf16) --------------------------
// grid: ceil(n/16) blocks, 256 threads (8 waves). Block computes a 16xD panel.
// Wave w owns column tiles {32w, 32w+16}. B fragments stream from global Wt.
__global__ void __launch_bounds__(256)
gat_gemm_h(const float* __restrict__ X, const __bf16* __restrict__ Wt,
           float* __restrict__ H, int n) {
  __shared__ alignas(16) __bf16 As[16 * AP];   // 16 rows x 256 K (bf16, padded)

  const int tid     = threadIdx.x;
  const int rowBase = blockIdx.x * 16;
  const int wave    = tid >> 5;
  const int lane    = tid & 31;
  const int lhalf   = lane >> 4;   // 0: lanes 0-15, 1: lanes 16-31
  const int lrow    = lane & 15;

  // Stage the 16x256 A panel as bf16: float4 coalesced reads, uint2 stores.
#pragma unroll
  for (int it = 0; it < 4; ++it) {
    int i = (it * 256 + tid) * 4;          // flat float index, 4-aligned
    int r = i >> 8, c = i & (D - 1);
    int gr = rowBase + r;
    float4 f = make_float4(0.f, 0.f, 0.f, 0.f);
    if (gr < n) f = *(const float4*)&X[(size_t)gr * D + c];
    Pack4 p;
    p.h[0] = (__bf16)f.x; p.h[1] = (__bf16)f.y;
    p.h[2] = (__bf16)f.z; p.h[3] = (__bf16)f.w;
    *(uint2*)&As[r * AP + c] = p.u;
  }
  __syncthreads();

  const int colBase0 = wave * 32;
  const int colBase1 = wave * 32 + 16;
  const __bf16* wt0 = &Wt[(size_t)(colBase0 + lrow) * D + lhalf * 16];
  const __bf16* wt1 = &Wt[(size_t)(colBase1 + lrow) * D + lhalf * 16];

  v8f acc0 = {};
  v8f acc1 = {};

#pragma unroll
  for (int kk = 0; kk < D; kk += 32) {
    // A fragment: 16 contiguous k per lane half (k = 16*lhalf + s), matching B.
    Frag16 a;
    a.q[0] = *(const uint4*)&As[lrow * AP + kk + lhalf * 16];
    a.q[1] = *(const uint4*)&As[lrow * AP + kk + lhalf * 16 + 8];

    // B fragments: 32 contiguous bytes each from transposed Wt (global, L2).
    Frag16 b0, b1;
    b0.q[0] = *(const uint4*)(wt0 + kk);
    b0.q[1] = *(const uint4*)(wt0 + kk + 8);
    b1.q[0] = *(const uint4*)(wt1 + kk);
    b1.q[1] = *(const uint4*)(wt1 + kk + 8);

    acc0 = __builtin_amdgcn_wmma_f32_16x16x32_bf16(false, a.v, false, b0.v,
                                                   (short)0, acc0, false, false);
    acc1 = __builtin_amdgcn_wmma_f32_16x16x32_bf16(false, a.v, false, b1.v,
                                                   (short)0, acc1, false, false);
  }

  // C/D layout: VGPR r -> row M = r + 8*lhalf, col N = lrow.
#pragma unroll
  for (int r = 0; r < 8; ++r) {
    int gr = rowBase + r + lhalf * 8;
    if (gr < n) {
      H[(size_t)gr * D + colBase0 + lrow] = acc0[r];
      H[(size_t)gr * D + colBase1 + lrow] = acc1[r];
    }
  }
}

// ------------------------- per-node attention scores -----------------------
// One wave per node: s1 = H[i].a[:D], s2 = H[i].a[D:].
__global__ void gat_scores(const float* __restrict__ H, const float* __restrict__ a,
                           float* __restrict__ s1, float* __restrict__ s2, int n) {
  int node = blockIdx.x * 8 + (threadIdx.x >> 5);
  int lane = threadIdx.x & 31;
  if (node >= n) return;
  float acc1 = 0.0f, acc2 = 0.0f;
#pragma unroll
  for (int d = lane; d < D; d += 32) {
    float hv = H[(size_t)node * D + d];
    acc1 += hv * a[d];
    acc2 += hv * a[D + d];
  }
#pragma unroll
  for (int off = 16; off > 0; off >>= 1) {
    acc1 += __shfl_down(acc1, off, 32);
    acc2 += __shfl_down(acc2, off, 32);
  }
  if (lane == 0) { s1[node] = acc1; s2[node] = acc2; }
}

// ------------------------- column-sum of H (empty-row fallback) ------------
__global__ void col_mean_acc(const float* __restrict__ H, float* __restrict__ meanacc,
                             int n) {
  int d = threadIdx.x;                                  // 256 threads
  int rowsPerBlock = (n + gridDim.x - 1) / gridDim.x;
  int r0 = blockIdx.x * rowsPerBlock;
  int r1 = min(r0 + rowsPerBlock, n);
  float acc = 0.0f;
  for (int r = r0; r < r1; ++r) acc += H[(size_t)r * D + d];
  atomicAdd(&meanacc[d], acc);
}

// ------------------------- edge pass 1: scores + row max -------------------
__global__ void edge_max(const int* __restrict__ src, const int* __restrict__ tgt,
                         const float* __restrict__ s1, const float* __restrict__ s2,
                         float* __restrict__ ev, float* __restrict__ rowmax, int E) {
  int k = blockIdx.x * 256 + threadIdx.x;
  if (k >= E) return;
  int i = src[k], j = tgt[k];
  float v = s1[i] + s2[j];
  v = fmaxf(v, ALPHA * v);                // leaky_relu (ALPHA < 1)
  ev[k] = v;
  __hip_atomic_fetch_max(&rowmax[i], v, __ATOMIC_RELAXED, __HIP_MEMORY_SCOPE_AGENT);
}

// ------------------------- edge pass 2: exp + row sum ----------------------
__global__ void edge_sum(const int* __restrict__ src, const float* __restrict__ ev,
                         const float* __restrict__ rowmax, float* __restrict__ wv,
                         float* __restrict__ rowsum, int E) {
  int k = blockIdx.x * 256 + threadIdx.x;
  if (k >= E) return;
  int i = src[k];
  float w = __expf(ev[k] - rowmax[i]);
  wv[k] = w;
  atomicAdd(&rowsum[i], w);
}

// ------------------------- initialize output -------------------------------
// Rows with edges start at bias; empty rows get mean(H) + bias (uniform softmax).
__global__ void out_init(const float* __restrict__ bias, const float* __restrict__ meanacc,
                         const float* __restrict__ rowsum, float* __restrict__ out,
                         int n) {
  int idx = blockIdx.x * 256 + threadIdx.x;
  if (idx >= n * D) return;
  int i = idx >> 8, d = idx & (D - 1);
  float v = bias[d];
  if (rowsum[i] <= 0.0f) v += meanacc[d] * (1.0f / (float)n);
  out[idx] = v;
}

// ------------------------- sparse aggregation ------------------------------
// One wave per edge: out[src] += (w_k / rowsum[src]) * H[tgt].
__global__ void aggregate(const int* __restrict__ src, const int* __restrict__ tgt,
                          const float* __restrict__ wv, const float* __restrict__ rowsum,
                          const float* __restrict__ H, float* __restrict__ out, int E) {
  int k = blockIdx.x * 8 + (threadIdx.x >> 5);
  if (k >= E) return;
  int lane = threadIdx.x & 31;
  int i = src[k], j = tgt[k];
  float coeff = wv[k] / rowsum[i];
#pragma unroll
  for (int d = lane; d < D; d += 32)
    atomicAdd(&out[(size_t)i * D + d], coeff * H[(size_t)j * D + d]);
}

// ---------------------------------------------------------------------------
extern "C" void kernel_launch(void* const* d_in, const int* in_sizes, int n_in,
                              void* d_out, int out_size, void* d_ws, size_t ws_size,
                              hipStream_t stream) {
  const float* X    = (const float*)d_in[0];
  const int*   src  = (const int*)  d_in[1];
  const int*   tgt  = (const int*)  d_in[2];
  const float* W    = (const float*)d_in[3];
  const float* a    = (const float*)d_in[4];
  const float* bias = (const float*)d_in[5];
  float* out = (float*)d_out;

  const int n = in_sizes[0] / D;   // 10000
  const int E = in_sizes[1];       // 320000

  // Carve workspace (~13.2 MB total).
  char* ws = (char*)d_ws;
  auto carve = [&](size_t bytes) -> char* {
    char* p = ws;
    ws += (bytes + 255) & ~(size_t)255;
    return p;
  };
  float*  H       = (float*) carve((size_t)n * D * sizeof(float));
  __bf16* Wt      = (__bf16*)carve((size_t)D * D * sizeof(__bf16));
  float*  s1      = (float*) carve((size_t)n * sizeof(float));
  float*  s2      = (float*) carve((size_t)n * sizeof(float));
  float*  ev      = (float*) carve((size_t)E * sizeof(float));
  float*  wv      = (float*) carve((size_t)E * sizeof(float));
  float*  rowmax  = (float*) carve((size_t)n * sizeof(float));
  float*  rowsum  = (float*) carve((size_t)n * sizeof(float));
  float*  meanacc = (float*) carve(D * sizeof(float));

  init_buffers<<<(n + 255) / 256, 256, 0, stream>>>(rowmax, rowsum, meanacc, n);
  prep_w      <<<D,               256, 0, stream>>>(W, Wt);
  gat_gemm_h  <<<(n + 15) / 16,   256, 0, stream>>>(X, Wt, H, n);
  gat_scores  <<<(n + 7) / 8,     256, 0, stream>>>(H, a, s1, s2, n);
  col_mean_acc<<<40,              256, 0, stream>>>(H, meanacc, n);
  edge_max    <<<(E + 255) / 256, 256, 0, stream>>>(src, tgt, s1, s2, ev, rowmax, E);
  edge_sum    <<<(E + 255) / 256, 256, 0, stream>>>(src, ev, rowmax, wv, rowsum, E);
  out_init    <<<(n * D + 255) / 256, 256, 0, stream>>>(bias, meanacc, rowsum, out, n);
  aggregate   <<<(E + 7) / 8,     256, 0, stream>>>(src, tgt, wv, rowsum, H, out, E);
}
